// SSDCouplingNeck_3332894621761
// MI455X (gfx1250) — matrix operand order repaired and checked by reading, not compiled, not run on hardware
//
#include <hip/hip_runtime.h>
#include <hip/hip_bf16.h>

// ---------------------------------------------------------------------------
// SSDCouplingNeck for MI455X (gfx1250, wave32, WMMA)
//   - conv1x1 GEMMs:    v_wmma_f32_16x16x32_f16, async global->LDS staged B
//                       panel (ASYNCcnt), 4x register blocking, fused k+v pass
//   - attention logits: v_wmma_f32_16x16x4_f32  (K = HEAD_DIM = 4, exact fit)
//   - P x V^T:          v_wmma_f32_16x16x32_f16 with flash-style online softmax
// All per-level kernels are templated on L so strides become immediate offsets.
// ---------------------------------------------------------------------------

typedef __attribute__((ext_vector_type(16))) _Float16 v16h;
typedef __attribute__((ext_vector_type(8)))  _Float16 v8h;
typedef __attribute__((ext_vector_type(8)))  float    v8f;
typedef __attribute__((ext_vector_type(2)))  float    v2f;

typedef __attribute__((address_space(3))) _Float16 lds_h;

#define NBATCH 2
#define CC     256
#define NHEADS 64
#define HD     4
#define TT     32   // ds tokens

// ---------------- elementwise helpers ----------------

__global__ void k_f32_to_f16(const float* __restrict__ src, _Float16* __restrict__ dst, int n) {
  int i = blockIdx.x * blockDim.x + threadIdx.x;
  if (i < n) dst[i] = (_Float16)src[i];
}

__global__ void k_zero(float* __restrict__ p, int n) {
  int i = blockIdx.x * blockDim.x + threadIdx.x;
  if (i < n) p[i] = 0.f;
}

// (N,C,L) f32 -> (C, N*L) f16   grid: (ceil(L/256), NBATCH*CC)
template<int L>
__global__ void __launch_bounds__(256)
k_pack_x(const float* __restrict__ in, _Float16* __restrict__ X) {
  unsigned nc = blockIdx.y;                  // n*CC + c
  unsigned l = blockIdx.x * 256 + threadIdx.x;
  if (l >= (unsigned)L) return;
  unsigned n = nc >> 8, c = nc & 255u;
  X[(size_t)c * (NBATCH * L) + (size_t)n * L + l] = (_Float16)in[(size_t)nc * L + l];
}

// pooled[n*C+c] = mean over HW of ds[n][c][:]
__global__ void k_pool(const float* __restrict__ ds, float* __restrict__ pooled, int L) {
  int b = blockIdx.x;                        // n*CC + c
  __shared__ float red[64];
  float s = 0.f;
  for (int i = threadIdx.x; i < L; i += 64) s += ds[(size_t)b * L + i];
  red[threadIdx.x] = s;
  __syncthreads();
  for (int off = 32; off > 0; off >>= 1) {
    if (threadIdx.x < off) red[threadIdx.x] += red[threadIdx.x + off];
    __syncthreads();
  }
  if (threadIdx.x == 0) pooled[b] = red[0] / (float)L;
}

// key_tok / val_tok = pooled @ W^T + b  (val gated by sigmoid(ds_gate))
__global__ void k_tokens(const float* __restrict__ pooled,
                         const float* __restrict__ Wk, const float* __restrict__ bk,
                         const float* __restrict__ Wv, const float* __restrict__ bv,
                         const float* __restrict__ gatep,
                         float* __restrict__ ktok, float* __restrict__ vtok) {
  int i = blockIdx.x * blockDim.x + threadIdx.x;
  if (i >= NBATCH * 8192) return;
  int o = i % 8192, n = i / 8192;
  const float* pp = pooled + n * CC;
  float sk = bk[o], sv = bv[o];
  for (int c = 0; c < CC; c += 4) {
    float4 x = *(const float4*)(pp + c);
    float4 wk = *(const float4*)(Wk + (size_t)o * CC + c);
    float4 wv = *(const float4*)(Wv + (size_t)o * CC + c);
    sk += wk.x * x.x + wk.y * x.y + wk.z * x.z + wk.w * x.w;
    sv += wv.x * x.x + wv.y * x.y + wv.z * x.z + wv.w * x.w;
  }
  float g = 1.f / (1.f + __expf(-gatep[0]));
  ktok[i] = sk;
  vtok[i] = sv * g;
}

// ---------------- conv1x1 GEMM: Y(256 x NL) = W(256x256) * X(256 x NL) + b --
// Block = 8 waves. Block tile: 64 (o) x 128 (p). Each wave: four 16x16 tiles
// per weight matrix sharing one async-staged LDS B panel. TWO=true computes
// two weight matrices (k & v) against the same panel.
// mode 0: store Y f32. mode 1: fused residual store to d_out.

template<int L, bool TWO>
__global__ void __launch_bounds__(256)
k_gemm(const _Float16* __restrict__ W1, const _Float16* __restrict__ W2,
       const _Float16* __restrict__ X,
       const float* __restrict__ b1, const float* __restrict__ b2,
       float* __restrict__ Y1, float* __restrict__ Y2,
       int mode, const float* __restrict__ sp, float* __restrict__ outp) {
  constexpr int NL   = NBATCH * L;
  constexpr int LSH  = __builtin_ctz((unsigned)L);
  constexpr int BSTR = 136;                  // padded LDS row stride (halves)
  __shared__ __align__(16) _Float16 Bt[32 * BSTR];

  const int tid = threadIdx.x;
  const int lane = tid & 31, wave = tid >> 5;
  const int col = lane & 15, hi = lane >> 4;
  const int p0 = blockIdx.x * 128;           // block column base
  const int o0 = blockIdx.y * 64;            // block row base (4 tiles / wave)
  const int pw = p0 + wave * 16;             // wave column tile
  const int sr = tid >> 3;                   // staging row (0..31)
  const int sc = (tid & 7) * 16;             // staging col offset (halves)

  const unsigned lds0 = (unsigned)(size_t)(lds_h*)&Bt[sr * BSTR + sc];
  const unsigned lds1 = lds0 + 16;           // +8 halves

  v8f acc1[4], acc2[4];
  for (int r = 0; r < 4; r++)
    for (int v = 0; v < 8; v++) { acc1[r][v] = 0.f; acc2[r][v] = 0.f; }

  for (int k0 = 0; k0 < CC; k0 += 32) {
    // async DMA stage of X[k0..k0+31][p0..p0+127] into LDS (ASYNCcnt-tracked)
    const _Float16* xs = X + (size_t)(k0 + sr) * NL + p0 + sc;
    asm volatile("global_load_async_to_lds_b128 %0, %1, off"
                 :: "v"(lds0), "v"(xs) : "memory");
    asm volatile("global_load_async_to_lds_b128 %0, %1, off"
                 :: "v"(lds1), "v"(xs + 8) : "memory");
    if (k0 + 32 < CC) __builtin_prefetch(xs + (size_t)32 * NL, 0, 0);
    asm volatile("s_wait_asynccnt 0x0" ::: "memory");
    __syncthreads();

    // B fragment (32x16 f16): per-lane-half K split {0..7,16..23}/{8..15,24..31}
    v16h b;
    for (int e = 0; e < 16; e++) b[e] = Bt[((e < 8) ? (hi * 8 + e) : (16 + hi * 8 + e - 8)) * BSTR + wave * 16 + col];

    for (int r = 0; r < 4; r++) {
      // A (16x32 f16): row = col, per-lane-half K split {0..7,16..23}/{8..15,24..31}
      const _Float16* wr = W1 + (size_t)(o0 + r * 16 + col) * CC + k0 + hi * 8;
      v8h a0 = *(const v8h*)wr;
      v8h a1 = *(const v8h*)(wr + 16);
      v16h a;
      for (int j = 0; j < 8; j++) { a[j] = a0[j]; a[8 + j] = a1[j]; }
      acc1[r] = __builtin_amdgcn_wmma_f32_16x16x32_f16(false, a, false, b, (short)0, acc1[r], false, false);
      if (TWO) {
        const _Float16* wr2 = W2 + (size_t)(o0 + r * 16 + col) * CC + k0 + hi * 8;
        v8h c0 = *(const v8h*)wr2;
        v8h c1 = *(const v8h*)(wr2 + 16);
        v16h a2;
        for (int j = 0; j < 8; j++) { a2[j] = c0[j]; a2[8 + j] = c1[j]; }
        acc2[r] = __builtin_amdgcn_wmma_f32_16x16x32_f16(false, a2, false, b, (short)0, acc2[r], false, false);
      }
    }
    __syncthreads();
  }

  if (mode == 0) {
    for (int r = 0; r < 4; r++)
      for (int v = 0; v < 8; v++) {
        int o = o0 + r * 16 + v + 8 * hi;    // C/D layout: row = v + 8*(lane>=16)
        Y1[(size_t)o * NL + pw + col] = acc1[r][v] + b1[o];
        if (TWO) Y2[(size_t)o * NL + pw + col] = acc2[r][v] + b2[o];
      }
  } else {
    unsigned p = (unsigned)(pw + col);
    unsigned n = p >> LSH, l = p & (unsigned)(L - 1);
    for (int r = 0; r < 4; r++)
      for (int v = 0; v < 8; v++) {
        int o = o0 + r * 16 + v + 8 * hi;
        size_t idx = (size_t)n * CC * L + (size_t)o * L + l;
        outp[idx] = sp[idx] + acc1[r][v] + b1[o];
      }
  }
}

// ---------------- attention kernel -----------------------------------------
// grid: (L/64, HEADS, N); 4 waves / block, each wave owns a 16-query strip.
// Streams keys in 32-blocks: 2x f32 WMMA (logits, K=4), online softmax,
// P staged to LDS as f16, 1x f16 WMMA for P x V^T.

template<int L>
__global__ void __launch_bounds__(128)
k_attn(const float* __restrict__ Yq, const float* __restrict__ Yk,
       const float* __restrict__ Yv, const float* __restrict__ ktok,
       const float* __restrict__ vtok, _Float16* __restrict__ Ob,
       float* __restrict__ dsracc) {
  constexpr int NL = NBATCH * L;
  constexpr int M  = L + TT;
  int lane = threadIdx.x & 31;
  int wave = threadIdx.x >> 5;
  int hi   = lane >> 4;
  int col  = lane & 15;            // also the A-fragment row index
  int h = blockIdx.y, n = blockIdx.z;
  int l0 = (blockIdx.x * 4 + wave) * 16;

  __shared__ _Float16 Pl[4][16 * 32];
  _Float16* P = Pl[wave];

  const float scale = 0.5f;        // 1/sqrt(HEAD_DIM=4), folded into q

  // A fragment for logits (f32 16x4): lanes<16 hold K=0,1; lanes>=16 K=2,3
  v2f aq;
  {
    const float* qb = Yq + (size_t)(h * HD) * NL + (size_t)n * L + l0 + col;
    aq[0] = qb[(size_t)(2 * hi + 0) * NL] * scale;
    aq[1] = qb[(size_t)(2 * hi + 1) * NL] * scale;
  }

  float m8[8], s8[8], d8[8];
  for (int v = 0; v < 8; v++) { m8[v] = -3.0e38f; s8[v] = 0.f; d8[v] = 0.f; }
  v8f accp;
  for (int v = 0; v < 8; v++) accp[v] = 0.f;

  const float* Kb = Yk + (size_t)(h * HD) * NL + (size_t)n * L;
  const float* Vb = Yv + (size_t)(h * HD) * NL + (size_t)n * L;
  const float* Kt = ktok + (size_t)n * 8192 + h * HD * TT;
  const float* Vt = vtok + (size_t)n * 8192 + h * HD * TT;

  for (int m0 = 0; m0 < M; m0 += 32) {
    bool isDs = (m0 >= L);         // last 32-key block == ds tokens (M-L == 32)
    v8f S[2];
    for (int t = 0; t < 2; t++) {
      int mt = m0 + t * 16;
      v2f bk;
      if (mt < L) {
        const float* kp = Kb + mt + col;
        bk[0] = kp[(size_t)(2 * hi + 0) * NL];
        bk[1] = kp[(size_t)(2 * hi + 1) * NL];
      } else {
        const float* kp = Kt + (mt - L) + col;
        bk[0] = kp[(2 * hi + 0) * TT];
        bk[1] = kp[(2 * hi + 1) * TT];
      }
      v8f z; for (int v = 0; v < 8; v++) z[v] = 0.f;
      S[t] = __builtin_amdgcn_wmma_f32_16x16x4_f32(false, aq, false, bk, (short)0, z, false, false);
    }

    // online softmax over the 32-key block; row = v + 8*hi, cols across 16 lanes
    for (int v = 0; v < 8; v++) {
      float tmax = fmaxf(S[0][v], S[1][v]);
      for (int msk = 8; msk > 0; msk >>= 1) tmax = fmaxf(tmax, __shfl_xor(tmax, msk, 16));
      float mnew = fmaxf(m8[v], tmax);
      float corr = __expf(m8[v] - mnew);
      float p0v = __expf(S[0][v] - mnew);
      float p1v = __expf(S[1][v] - mnew);
      float rs = p0v + p1v;
      for (int msk = 8; msk > 0; msk >>= 1) rs += __shfl_xor(rs, msk, 16);
      s8[v] = s8[v] * corr + rs;
      d8[v] = d8[v] * corr + (isDs ? rs : 0.f);
      m8[v] = mnew;
      accp[v] *= corr;
      int row = v + 8 * hi;
      P[row * 32 + col]      = (_Float16)p0v;
      P[row * 32 + 16 + col] = (_Float16)p1v;
    }
    asm volatile("s_wait_dscnt 0" ::: "memory");

    // A fragment (P, 16x32 f16) reloaded from LDS in WMMA layout
    v16h ap, bv;
    for (int j = 0; j < 8; j++) {
      ap[j]     = P[col * 32 + hi * 16 + j];
      ap[8 + j] = P[col * 32 + hi * 16 + 8 + j];
    }
    // B fragment (V^T, 32x16 f16): only first HD=4 columns are real
    {
      int kb = hi * 16;
      if (col < HD) {
        const float* vp = isDs ? (Vt + col * TT + (m0 - L) + kb)
                               : (Vb + (size_t)col * NL + m0 + kb);
        for (int e = 0; e < 16; e += 4) {
          float4 f = *(const float4*)(vp + e);
          bv[e + 0] = (_Float16)f.x; bv[e + 1] = (_Float16)f.y;
          bv[e + 2] = (_Float16)f.z; bv[e + 3] = (_Float16)f.w;
        }
      } else {
        for (int e = 0; e < 16; e++) bv[e] = (_Float16)0.f;
      }
    }
    accp = __builtin_amdgcn_wmma_f32_16x16x32_f16(false, ap, false, bv, (short)0, accp, false, false);
  }

  // epilogue: normalize, store attn output as f16 (C x NL) for the o-conv
  if (col < HD) {
    for (int v = 0; v < 8; v++) {
      int row = v + 8 * hi;
      Ob[(size_t)(h * HD + col) * NL + (size_t)n * L + l0 + row] = (_Float16)(accp[v] / s8[v]);
    }
  }
  // ds ratio contribution: rows are replicated across the 16 lanes of a half
  if (col == 0) {
    float acc = 0.f;
    for (int v = 0; v < 8; v++) acc += d8[v] / s8[v];
    atomicAdd(&dsracc[n], acc);
  }
}

__global__ void k_fin(const float* __restrict__ acc, float* __restrict__ o) {
  int i = threadIdx.x;
  if (i >= 8) return;
  const int Ls[4] = {4096, 1024, 256, 64};
  o[i] = acc[i] / (float)(NHEADS * Ls[i >> 1]);   // mean over heads x queries
}

// ---------------- host launcher --------------------------------------------

struct LevelArgs {
  const float *sp, *inv, *ds;
  const _Float16 *Wq16, *Wk16, *Wv16, *Wo16;
  const float *bq, *bk, *bv, *bo;
  const float *Wdsk, *bdsk, *Wdsv, *bdsv, *gate;
  float* dsracc;
  float* outl;
};

template<int L>
static void run_level(const LevelArgs& A, char* base, size_t& off, hipStream_t stream) {
  constexpr int NL = NBATCH * L;
  auto alloc = [&](size_t bytes) -> char* {
    off = (off + 255) & ~(size_t)255;
    char* r = base + off;
    off += bytes;
    return r;
  };
  _Float16* Xsp = (_Float16*)alloc((size_t)CC * NL * 2);
  _Float16* Xin = (_Float16*)alloc((size_t)CC * NL * 2);
  float* Yq = (float*)alloc((size_t)CC * NL * 4);
  float* Yk = (float*)alloc((size_t)CC * NL * 4);
  float* Yv = (float*)alloc((size_t)CC * NL * 4);
  float* pooled = (float*)alloc((size_t)NBATCH * CC * 4);
  float* ktok = (float*)alloc((size_t)NBATCH * 8192 * 4);
  float* vtok = (float*)alloc((size_t)NBATCH * 8192 * 4);
  _Float16* Ob = (_Float16*)alloc((size_t)CC * NL * 2);

  dim3 gp((L + 255) / 256, NBATCH * CC);
  k_pack_x<L><<<gp, 256, 0, stream>>>(A.sp,  Xsp);
  k_pack_x<L><<<gp, 256, 0, stream>>>(A.inv, Xin);
  k_pool<<<NBATCH * CC, 64, 0, stream>>>(A.ds, pooled, L);
  k_tokens<<<(NBATCH * 8192 + 255) / 256, 256, 0, stream>>>(
      pooled, A.Wdsk, A.bdsk, A.Wdsv, A.bdsv, A.gate, ktok, vtok);

  dim3 gg(NL / 128, 4);
  k_gemm<L, false><<<gg, 256, 0, stream>>>(A.Wq16, nullptr, Xsp, A.bq, nullptr,
                                           Yq, nullptr, 0, nullptr, nullptr);
  k_gemm<L, true><<<gg, 256, 0, stream>>>(A.Wk16, A.Wv16, Xin, A.bk, A.bv,
                                          Yk, Yv, 0, nullptr, nullptr);

  dim3 ga(L / 64, NHEADS, NBATCH);
  k_attn<L><<<ga, 128, 0, stream>>>(Yq, Yk, Yv, ktok, vtok, Ob, A.dsracc);

  k_gemm<L, false><<<gg, 256, 0, stream>>>(A.Wo16, nullptr, Ob, A.bo, nullptr,
                                           nullptr, nullptr, 1, A.sp, A.outl);
}

extern "C" void kernel_launch(void* const* d_in, const int* in_sizes, int n_in,
                              void* d_out, int out_size, void* d_ws, size_t ws_size,
                              hipStream_t stream) {
  (void)in_sizes; (void)n_in; (void)out_size; (void)ws_size;
  static const int HWs[4] = {64, 32, 16, 8};

  const float* Wsrc[4] = {(const float*)d_in[12], (const float*)d_in[14],
                          (const float*)d_in[16], (const float*)d_in[18]};
  const float* bq  = (const float*)d_in[13];
  const float* bkk = (const float*)d_in[15];
  const float* bvv = (const float*)d_in[17];
  const float* bo  = (const float*)d_in[19];
  const float* Wdsk = (const float*)d_in[20];
  const float* bdsk = (const float*)d_in[21];
  const float* Wdsv = (const float*)d_in[22];
  const float* bdsv = (const float*)d_in[23];
  const float* gate = (const float*)d_in[24];

  char* base = (char*)d_ws;
  size_t off = 0;
  auto alloc = [&](size_t bytes) -> char* {
    off = (off + 255) & ~(size_t)255;
    char* r = base + off;
    off += bytes;
    return r;
  };

  // f16 weights (4 tensors of shape (4,256,256))
  _Float16* W16[4];
  for (int i = 0; i < 4; i++) {
    W16[i] = (_Float16*)alloc((size_t)4 * 65536 * 2);
    k_f32_to_f16<<<(4 * 65536 + 255) / 256, 256, 0, stream>>>(Wsrc[i], W16[i], 4 * 65536);
  }

  float* dsracc = (float*)alloc(8 * sizeof(float));
  k_zero<<<1, 32, 0, stream>>>(dsracc, 8);

  size_t out_off = 0;
  for (int lvl = 0; lvl < 4; lvl++) {
    int HW = HWs[lvl], L = HW * HW;
    LevelArgs A;
    A.sp  = (const float*)d_in[lvl];
    A.inv = (const float*)d_in[4 + lvl];
    A.ds  = (const float*)d_in[8 + lvl];
    A.Wq16 = W16[0] + (size_t)lvl * 65536;
    A.Wk16 = W16[1] + (size_t)lvl * 65536;
    A.Wv16 = W16[2] + (size_t)lvl * 65536;
    A.Wo16 = W16[3] + (size_t)lvl * 65536;
    A.bq = bq + lvl * 256; A.bk = bkk + lvl * 256;
    A.bv = bvv + lvl * 256; A.bo = bo + lvl * 256;
    A.Wdsk = Wdsk + (size_t)lvl * 8192 * 256; A.bdsk = bdsk + (size_t)lvl * 8192;
    A.Wdsv = Wdsv + (size_t)lvl * 8192 * 256; A.bdsv = bdsv + (size_t)lvl * 8192;
    A.gate = gate + lvl;
    A.dsracc = dsracc;
    A.outl = (float*)d_out + out_off;

    switch (lvl) {
      case 0: run_level<4096>(A, base, off, stream); break;
      case 1: run_level<1024>(A, base, off, stream); break;
      case 2: run_level<256>(A, base, off, stream); break;
      default: run_level<64>(A, base, off, stream); break;
    }
    out_off += (size_t)NBATCH * CC * L;
  }

  k_fin<<<1, 8, 0, stream>>>(dsracc, (float*)d_out + out_off);
}
